// SpatialAttentionLayer_30253749633179
// MI455X (gfx1250) — matrix operand order repaired and checked by reading, not compiled, not run on hardware
//
#include <hip/hip_runtime.h>
#include <hip/hip_bf16.h>
#include <math.h>

// ---------------------------------------------------------------------------
// SpatialAttentionLayer for MI455X (gfx1250, wave32, WMMA).
// G=2 items per 256-thread block (8 waves): M=64 rows amortizes every weight
// fetch over 2 items, and B-fragments are reused across all 4 m-tiles, so each
// weight element is read from L2 exactly once per workgroup (~6.5MB/wg for 2
// items -> ~13 GB total, vs 53 GB before). All GEMMs use
// v_wmma_f32_16x16x32_bf16 with f32 accumulation; weights pre-converted to
// bf16 transposed (N-major) in d_ws so B-fragments are two contiguous 8-elem k runs per lane.
// Attention is processed in head-pair passes so 2-item buffers fit in LDS.
// Output accumulates scale contributions directly into global out (no races:
// disjoint (m,n) ownership; out = src written first).
// ---------------------------------------------------------------------------

typedef __attribute__((ext_vector_type(16))) __bf16 v16bf;
typedef __attribute__((ext_vector_type(8)))  __bf16 v8bf;
typedef __attribute__((ext_vector_type(8)))  float  v8f;

#define C_DIM   512
#define HS      64
#define NJ      25
#define NBT     4096
#define ATTN_SC 0.125f          // 64^-0.5
#define SQ      520             // [64][SQ] bf16 row stride (qn/lf/kn)
#define SQH     136             // [64][SQH] bf16 (head-pair q/k, 128 cols)
#define SVT     72              // vT2 row stride ([128 ch][SVT])
#define SH      1032            // pooling hidden row stride

// LDS arena (bytes)
#define P_QN    0               // qn [64][SQ] bf16 = 66560 (persistent)
#define P_LF    66560           // lf [64][SQ] bf16 = 66560 (persistent)
#define P_G     133120          // gamma[512] f32
#define P_B     135168          // beta [512] f32
#define OFF_R   137216          // reusable region (190464 available)
// attention phase (per scale / per head-pair)
#define R_KN    0               // kn  [64][SQ]   bf16 (66560)
#define R_Q2    66560           // q2  [64][SQH]  bf16 (17408)
#define R_K2    83968           // k2  [64][SQH]  bf16 (17408)
#define R_VT    101376          // vT2 [128][SVT] bf16 (18432)
#define R_LG    119808          // lg  [2][64][32] f32 (16384)
#define R_PB    136192          // pb  [2][64][32] bf16 (8192)  -> end 144384
// pooling phase (aliases attention buffers)
#define R_H     0               // h [64][SH] bf16 (132096)
#define R_SM    132096          // Sm [2][32][16] f32 (4096)
#define R_LN2   136192          // lnew [2][12][512] f32 (49152) -> end 185344
#define ARENA_BYTES 322560      // 137216 + 185344 < 327680

// ---- WMMA fragment loaders (ISA §7.12.2 layouts) ---------------------------
__device__ __forceinline__ v16bf afrag(const __bf16* base, int stride,
                                       int m0, int k0, int lane) {
  int row = m0 + (lane & 15);
  int kk  = k0 + ((lane >> 4) << 3);
  const __bf16* p = base + (size_t)row * stride + kk;
  v8bf lo = *(const v8bf*)p;
  v8bf hi = *(const v8bf*)(p + 16);
  v16bf r;
#pragma unroll
  for (int i = 0; i < 8; ++i) { r[i] = lo[i]; r[i + 8] = hi[i]; }
  return r;
}
__device__ __forceinline__ v16bf bfrag(const __bf16* baseT, int stride,
                                       int n0, int k0, int lane) {
  int row = n0 + (lane & 15);
  int kk  = k0 + ((lane >> 4) << 3);
  const __bf16* p = baseT + (size_t)row * stride + kk;
  v8bf lo = *(const v8bf*)p;
  v8bf hi = *(const v8bf*)(p + 16);
  v16bf r;
#pragma unroll
  for (int i = 0; i < 8; ++i) { r[i] = lo[i]; r[i + 8] = hi[i]; }
  return r;
}
#define WMMA_BF16(a, b, c) \
  __builtin_amdgcn_wmma_f32_16x16x32_bf16(false, (a), false, (b), (short)0, (c), false, false)

__device__ __forceinline__ v8f mmtile(const __bf16* A, int sa, int m0,
                                      const __bf16* Bt, int sb, int n0,
                                      int ksteps, int lane) {
  v8f acc = {};
  for (int ks = 0; ks < ksteps; ++ks) {
    v16bf a = afrag(A, sa, m0, ks * 32, lane);
    v16bf b = bfrag(Bt, sb, n0, ks * 32, lane);
    acc = WMMA_BF16(a, b, acc);
  }
  return acc;
}

// LN of rows (item*32 + r), r < Ni, reading bf16 lf, writing bf16 dst
__device__ __forceinline__ void layernorm_g2(const __bf16* x, __bf16* dst,
                                             int Ni, const float* g,
                                             const float* b, int wave, int lane) {
  for (int idx = wave; idx < 2 * Ni; idx += 8) {
    int item = idx / Ni, r = idx - item * Ni;
    const __bf16* xr = x + (item * 32 + r) * SQ;
    float s = 0.f, s2 = 0.f;
    for (int c = lane; c < C_DIM; c += 32) { float v = (float)xr[c]; s += v; s2 += v * v; }
#pragma unroll
    for (int o = 16; o > 0; o >>= 1) {
      s  += __shfl_xor(s,  o, 32);
      s2 += __shfl_xor(s2, o, 32);
    }
    float mu   = s  * (1.f / C_DIM);
    float var  = s2 * (1.f / C_DIM) - mu * mu;
    float rstd = rsqrtf(var + 1e-5f);
    for (int c = lane; c < C_DIM; c += 32)
      dst[(item * 32 + r) * SQ + c] = (__bf16)(((float)xr[c] - mu) * rstd * g[c] + b[c]);
  }
}

// ---- weight prep: f32 [m][K][N] -> bf16 [m][N][K] --------------------------
__global__ void __launch_bounds__(256)
transpose_bf16_kernel(const float* __restrict__ w, __bf16* __restrict__ o,
                      int K, int N, int total) {
  int idx = blockIdx.x * blockDim.x + threadIdx.x;
  if (idx >= total) return;
  int per = K * N;
  int m = idx / per;
  int r = idx - m * per;
  int n = r / K;
  int k = r - n * K;
  o[idx] = (__bf16)w[(size_t)m * per + (size_t)k * N + n];
}

// ---- main fused kernel (2 items per block) ---------------------------------
__global__ void __launch_bounds__(256)
spatial_attn_kernel(const float* __restrict__ src,
                    const float* __restrict__ ln_g, const float* __restrict__ ln_b,
                    const __bf16* __restrict__ WqT, const __bf16* __restrict__ WkT,
                    const __bf16* __restrict__ WvT,
                    const __bf16* __restrict__ f1T0, const __bf16* __restrict__ f1T1,
                    const float* __restrict__ f1b0, const float* __restrict__ f1b1,
                    const float* __restrict__ f2w0, const float* __restrict__ f2b0,
                    const float* __restrict__ f2w1, const float* __restrict__ f2b1,
                    float* __restrict__ out) {
  extern __shared__ char smem[];
  __bf16* qn  = (__bf16*)(smem + P_QN);
  __bf16* lf  = (__bf16*)(smem + P_LF);
  float*  sg  = (float*)(smem + P_G);
  float*  sbt = (float*)(smem + P_B);
  char*   R   = smem + OFF_R;
  __bf16* kn  = (__bf16*)(R + R_KN);
  __bf16* q2  = (__bf16*)(R + R_Q2);
  __bf16* k2  = (__bf16*)(R + R_K2);
  __bf16* vT2 = (__bf16*)(R + R_VT);
  float*  lg  = (float*)(R + R_LG);
  __bf16* pb  = (__bf16*)(R + R_PB);
  __bf16* hb  = (__bf16*)(R + R_H);
  float*  Sm  = (float*)(R + R_SM);
  float*  ln2 = (float*)(R + R_LN2);

  const int tid  = threadIdx.x;
  const int lane = tid & 31;
  const int wave = tid >> 5;
  const int bt0  = blockIdx.x * 2;
  const float* xs[2] = { src + (size_t)bt0 * NJ * C_DIM,
                         src + (size_t)(bt0 + 1) * NJ * C_DIM };
  float* op[2] = { out + (size_t)bt0 * NJ * C_DIM,
                   out + (size_t)(bt0 + 1) * NJ * C_DIM };

  // init: lf (bf16, zero-padded), LN params, out = src
  for (int i = tid; i < 64 * SQ; i += 256) {
    int r = i / SQ, c = i - r * SQ;
    int item = r >> 5, m = r & 31;
    float v = (m < NJ && c < C_DIM) ? xs[item][m * C_DIM + c] : 0.f;
    lf[i] = (__bf16)v;
  }
  for (int i = tid; i < C_DIM; i += 256) { sg[i] = ln_g[i]; sbt[i] = ln_b[i]; }
  for (int i = tid; i < 2 * NJ * C_DIM; i += 256) {
    int item = i / (NJ * C_DIM), r = i - item * NJ * C_DIM;
    op[item][r] = xs[item][r];
  }
  // zero qn pad rows
  for (int i = tid; i < 2 * (32 - NJ) * SQ; i += 256) {
    int item = i / ((32 - NJ) * SQ), rem = i - item * (32 - NJ) * SQ;
    qn[(item * 32 + NJ) * SQ + rem] = (__bf16)0.f;
  }
  __syncthreads();
  layernorm_g2(lf, qn, NJ, sg, sbt, wave, lane);   // qn = LN(src), constant
  __syncthreads();

  const int scales[3] = {25, 12, 7};
  for (int sc = 0; sc < 3; ++sc) {
    const int Ni = scales[sc];
    const __bf16* Wq = WqT + (size_t)sc * C_DIM * C_DIM;
    const __bf16* Wk = WkT + (size_t)sc * C_DIM * C_DIM;
    const __bf16* Wv = WvT + (size_t)sc * C_DIM * C_DIM;

    // kn = LN(l_feat), pad rows zero
    layernorm_g2(lf, kn, Ni, sg, sbt, wave, lane);
    for (int i = tid; i < 2 * (32 - Ni) * SQ; i += 256) {
      int item = i / ((32 - Ni) * SQ), rem = i - item * (32 - Ni) * SQ;
      kn[(item * 32 + Ni) * SQ + rem] = (__bf16)0.f;
    }
    __syncthreads();

    for (int hp = 0; hp < 4; ++hp) {         // head pairs (2 heads, 128 channels)
      // Q/K/V projections: M=64 (4 m-tiles), N=128 (8 n-tiles), K=512.
      // B fragment loaded once per (n-tile, k-step), reused for 4 m-tiles.
      for (int combo = wave; combo < 24; combo += 8) {
        int which = combo >> 3, nt = combo & 7;
        const __bf16* A  = (which == 0) ? qn : kn;
        const __bf16* Bt = (which == 0) ? Wq : (which == 1) ? Wk : Wv;
        int n0 = hp * 128 + nt * 16;
        v8f a0 = {}, a1 = {}, a2 = {}, a3 = {};
        for (int ks = 0; ks < 16; ++ks) {
          v16bf bf = bfrag(Bt, C_DIM, n0, ks * 32, lane);
          a0 = WMMA_BF16(afrag(A, SQ,  0, ks * 32, lane), bf, a0);
          a1 = WMMA_BF16(afrag(A, SQ, 16, ks * 32, lane), bf, a1);
          a2 = WMMA_BF16(afrag(A, SQ, 32, ks * 32, lane), bf, a2);
          a3 = WMMA_BF16(afrag(A, SQ, 48, ks * 32, lane), bf, a3);
        }
        int nl = nt * 16 + (lane & 15);            // channel in head-pair [0,128)
        int mb = (lane >> 4) << 3;
        v8f accs[4] = { a0, a1, a2, a3 };
#pragma unroll
        for (int mt = 0; mt < 4; ++mt) {
#pragma unroll
          for (int r2 = 0; r2 < 8; ++r2) {
            int m = mt * 16 + mb + r2;
            float v = accs[mt][r2];
            if (which == 0)      q2[m * SQH + nl] = (__bf16)(v * ATTN_SC);
            else if (which == 1) k2[m * SQH + nl] = (__bf16)v;
            else                 vT2[nl * SVT + m] = (__bf16)v;   // transposed
          }
        }
      }
      __syncthreads();

      // logits: per head (hh) and item (it): q (32x64) @ k^T, K=64
      for (int t = wave; t < 16; t += 8) {
        int hh = t >> 3, it = (t >> 2) & 1, mt = (t >> 1) & 1, nt = t & 1;
        v8f acc = mmtile(q2 + hh * HS, SQH, it * 32 + mt * 16,
                         k2 + hh * HS, SQH, it * 32 + nt * 16, 2, lane);
        int n  = nt * 16 + (lane & 15);            // local key index [0,32)
        int mb = (lane >> 4) << 3;
#pragma unroll
        for (int r2 = 0; r2 < 8; ++r2)
          lg[(hh * 64 + it * 32 + mt * 16 + mb + r2) * 32 + n] = acc[r2];
      }
      __syncthreads();

      // softmax: one thread per (head, item, row)
      if (tid < 128) {
        int hh = tid >> 6, rg = tid & 63;
        int m = rg & 31;
        const float* row = lg + (hh * 64 + rg) * 32;
        __bf16* prow = pb + (hh * 64 + rg) * 32;
        if (m < NJ) {
          float mx = -3.0e38f;
          for (int n = 0; n < Ni; ++n) mx = fmaxf(mx, row[n]);
          float ssum = 0.f;
          for (int n = 0; n < Ni; ++n) ssum += __expf(row[n] - mx);
          float inv = 1.f / ssum;
          for (int n = 0; n < 32; ++n)
            prow[n] = (n < Ni) ? (__bf16)(__expf(row[n] - mx) * inv) : (__bf16)0.f;
        } else {
          for (int n = 0; n < 32; ++n) prow[n] = (__bf16)0.f;
        }
      }
      __syncthreads();

      // o = p (32x32) @ v (32x64); accumulate o/3 into global out
      for (int t = wave; t < 32; t += 8) {
        int hh = t >> 4, it = (t >> 3) & 1, mt = (t >> 2) & 1, nt = t & 3;
        v8f acc = mmtile(pb + (hh * 64 + it * 32) * 32, 32, mt * 16,
                         vT2 + it * 32, SVT, hh * HS + nt * 16, 1, lane);
        int nglob = hp * 128 + hh * HS + nt * 16 + (lane & 15);
        int mb = (lane >> 4) << 3;
        float* o_it = op[it];
#pragma unroll
        for (int r2 = 0; r2 < 8; ++r2) {
          int m = mt * 16 + mb + r2;
          if (m < NJ) o_it[m * C_DIM + nglob] += acc[r2] * (1.f / 3.f);
        }
      }
      __syncthreads();
    }

    // pooling MLP (both items fused, M=64): l_feat -> (No x 512) per item
    if (sc < 2) {
      const int No = scales[sc + 1];
      const __bf16* f1  = (sc == 0) ? f1T0 : f1T1;
      const float*  f1b = (sc == 0) ? f1b0 : f1b1;
      const float*  f2w = (sc == 0) ? f2w0 : f2w1;
      const float*  f2b = (sc == 0) ? f2b0 : f2b1;

      // h = gelu(lf @ fc1 + b): N=1024 (64 n-tiles), K=512; B reused over 4 m-tiles
      for (int nt = wave; nt < 64; nt += 8) {
        v8f a0 = {}, a1 = {}, a2 = {}, a3 = {};
        for (int ks = 0; ks < 16; ++ks) {
          v16bf bf = bfrag(f1, C_DIM, nt * 16, ks * 32, lane);
          a0 = WMMA_BF16(afrag(lf, SQ,  0, ks * 32, lane), bf, a0);
          a1 = WMMA_BF16(afrag(lf, SQ, 16, ks * 32, lane), bf, a1);
          a2 = WMMA_BF16(afrag(lf, SQ, 32, ks * 32, lane), bf, a2);
          a3 = WMMA_BF16(afrag(lf, SQ, 48, ks * 32, lane), bf, a3);
        }
        int n = nt * 16 + (lane & 15);
        float bias = f1b[n];
        int mb = (lane >> 4) << 3;
        v8f accs[4] = { a0, a1, a2, a3 };
#pragma unroll
        for (int mt = 0; mt < 4; ++mt) {
#pragma unroll
          for (int r2 = 0; r2 < 8; ++r2) {
            float x = accs[mt][r2] + bias;
            hb[(mt * 16 + mb + r2) * SH + n] =
                (__bf16)(0.5f * x * (1.f + erff(x * 0.70710678118f)));
          }
        }
      }
      __syncthreads();
      // S logits: per item (Ni x 1024) @ (1024 x No) — tiny N, VALU dots
      for (int i = tid; i < 2 * Ni * No; i += 256) {
        int item = i / (Ni * No), rem = i - item * Ni * No;
        int n = rem / No, s2 = rem - n * No;
        float a = f2b[s2];
        const __bf16* hr = hb + (item * 32 + n) * SH;
        for (int k = 0; k < 2 * C_DIM; ++k) a += (float)hr[k] * f2w[k * No + s2];
        Sm[(item * 32 + n) * 16 + s2] = a;
      }
      __syncthreads();
      if (tid < 2 * Ni) {
        int item = tid / Ni, n = tid - item * Ni;
        float* rr = Sm + (item * 32 + n) * 16;
        float mx = -3.0e38f;
        for (int s2 = 0; s2 < No; ++s2) mx = fmaxf(mx, rr[s2]);
        float ssum = 0.f;
        for (int s2 = 0; s2 < No; ++s2) { float e = __expf(rr[s2] - mx); rr[s2] = e; ssum += e; }
        float inv = 1.f / ssum;
        for (int s2 = 0; s2 < No; ++s2) rr[s2] *= inv;
      }
      __syncthreads();
      // l_new = S^T @ l_feat (per item, No x 512)
      for (int i = tid; i < 2 * No * C_DIM; i += 256) {
        int item = i / (No * C_DIM), rem = i - item * No * C_DIM;
        int s2 = rem / C_DIM, c = rem - s2 * C_DIM;
        float a = 0.f;
        for (int n = 0; n < Ni; ++n)
          a += Sm[(item * 32 + n) * 16 + s2] * (float)lf[(item * 32 + n) * SQ + c];
        ln2[i] = a;
      }
      __syncthreads();
      // copy back (bf16) and zero rows [No, NJ)
      for (int i = tid; i < 2 * No * C_DIM; i += 256) {
        int item = i / (No * C_DIM), rem = i - item * No * C_DIM;
        int s2 = rem / C_DIM, c = rem - s2 * C_DIM;
        lf[(item * 32 + s2) * SQ + c] = (__bf16)ln2[i];
      }
      for (int i = tid; i < 2 * (NJ - No) * C_DIM; i += 256) {
        int item = i / ((NJ - No) * C_DIM), rem = i - item * (NJ - No) * C_DIM;
        int r = rem / C_DIM, c = rem - r * C_DIM;
        lf[(item * 32 + No + r) * SQ + c] = (__bf16)0.f;
      }
      __syncthreads();
    }
  }
}

// ---------------------------------------------------------------------------
extern "C" void kernel_launch(void* const* d_in, const int* in_sizes, int n_in,
                              void* d_out, int out_size, void* d_ws, size_t ws_size,
                              hipStream_t stream) {
  (void)in_sizes; (void)n_in; (void)out_size; (void)ws_size;
  const float* src  = (const float*)d_in[0];
  // d_in[1] spatial_relation: unused by the reference output math
  const float* ln_g = (const float*)d_in[2];
  const float* ln_b = (const float*)d_in[3];
  const float* Wq   = (const float*)d_in[4];
  const float* Wk   = (const float*)d_in[5];
  const float* Wv   = (const float*)d_in[6];
  const float* f1w0 = (const float*)d_in[7];
  const float* f1b0 = (const float*)d_in[8];
  const float* f2w0 = (const float*)d_in[9];
  const float* f2b0 = (const float*)d_in[10];
  const float* f1w1 = (const float*)d_in[11];
  const float* f1b1 = (const float*)d_in[12];
  const float* f2w1 = (const float*)d_in[13];
  const float* f2b1 = (const float*)d_in[14];
  float* out = (float*)d_out;

  __bf16* ws   = (__bf16*)d_ws;
  __bf16* WqT  = ws;
  __bf16* WkT  = WqT + (size_t)3 * 512 * 512;
  __bf16* WvT  = WkT + (size_t)3 * 512 * 512;
  __bf16* f1T0 = WvT + (size_t)3 * 512 * 512;
  __bf16* f1T1 = f1T0 + (size_t)1024 * 512;

  int tQ = 3 * 512 * 512;
  transpose_bf16_kernel<<<(tQ + 255) / 256, 256, 0, stream>>>(Wq, WqT, 512, 512, tQ);
  transpose_bf16_kernel<<<(tQ + 255) / 256, 256, 0, stream>>>(Wk, WkT, 512, 512, tQ);
  transpose_bf16_kernel<<<(tQ + 255) / 256, 256, 0, stream>>>(Wv, WvT, 512, 512, tQ);
  int tF = 512 * 1024;
  transpose_bf16_kernel<<<(tF + 255) / 256, 256, 0, stream>>>(f1w0, f1T0, 512, 1024, tF);
  transpose_bf16_kernel<<<(tF + 255) / 256, 256, 0, stream>>>(f1w1, f1T1, 512, 1024, tF);

  spatial_attn_kernel<<<NBT / 2, 256, ARENA_BYTES, stream>>>(
      src, ln_g, ln_b, WqT, WkT, WvT, f1T0, f1T1,
      f1b0, f1b1, f2w0, f2b0, f2w1, f2b1, out);
}
